// RotaryEmbeddingTransformerLayer_36919538876926
// MI455X (gfx1250) — matrix operand order repaired and checked by reading, not compiled, not run on hardware
//
#include <hip/hip_runtime.h>
#include <hip/hip_bf16.h>
#include <stdint.h>

// ---------------------------------------------------------------------------
// MI455X / gfx1250, wave32.  bf16 WMMA (v_wmma_f32_16x16x32_bf16) everywhere.
// ---------------------------------------------------------------------------

typedef __attribute__((ext_vector_type(16))) __bf16 v16bf;
typedef __attribute__((ext_vector_type(8)))  __bf16 v8bf;
typedef __attribute__((ext_vector_type(8)))  float  v8f;

#define D_MODEL 1024
#define N_HEADS 16
#define HEAD_DIM 64
#define D_FF 4096
#define SEQ 2048
#define BATCH 2
#define NTOK (BATCH * SEQ)   // 4096 token rows

__device__ __forceinline__ __bf16 f2bf(float f) {
    union { float f; uint32_t u; } a; a.f = f;
    uint32_t u = (a.u + 0x7FFFu + ((a.u >> 16) & 1u)) >> 16;  // RNE
    union { unsigned short s; __bf16 b; } o; o.s = (unsigned short)u;
    return o.b;
}

// A-operand fragment (16x32 bf16): lane m = lane&15; half = lane>>4 selects
// k = half*8 + {0..7}  and  16 + half*8 + {0..7}   (ISA 7.12.2)
__device__ __forceinline__ v16bf ld_fragA(const __bf16* rowp, int half) {
    v8bf lo = *(const v8bf*)(rowp + half * 8);
    v8bf hi = *(const v8bf*)(rowp + 16 + half * 8);
    v16bf r;
#pragma unroll
    for (int i = 0; i < 8; ++i) { r[i] = lo[i]; r[8 + i] = hi[i]; }
    return r;
}

// B-operand fragment (32x16 bf16, stored [N][K] row-major): lane n = lane&15;
// half selects k = half*8 + {0..7} and 16 + half*8 + {0..7} (same as A)
__device__ __forceinline__ v16bf ld_fragB(const __bf16* rowp, int half) {
    v8bf lo = *(const v8bf*)(rowp + half * 8);
    v8bf hi = *(const v8bf*)(rowp + 16 + half * 8);
    v16bf r;
#pragma unroll
    for (int i = 0; i < 8; ++i) { r[i] = lo[i]; r[8 + i] = hi[i]; }
    return r;
}

__device__ __forceinline__ v8f wmma_bf16(v16bf a, v16bf b, v8f c) {
    return __builtin_amdgcn_wmma_f32_16x16x32_bf16(false, a, false, b,
                                                   (short)0, c, false, false);
}

// ---------------------------------------------------------------------------
// fp32 -> bf16 weight conversion
// ---------------------------------------------------------------------------
__global__ void to_bf16_k(const float* __restrict__ in, __bf16* __restrict__ out, int n) {
    int i = blockIdx.x * blockDim.x + threadIdx.x;
    if (i < n) out[i] = f2bf(in[i]);
}

// ---------------------------------------------------------------------------
// LayerNorm over 1024, bf16 out.  One 256-thread block (8 waves) per row.
// ---------------------------------------------------------------------------
__device__ __forceinline__ float block_sum256(float v, float* sh) {
#pragma unroll
    for (int o = 16; o > 0; o >>= 1) v += __shfl_xor(v, o);
    int wave = threadIdx.x >> 5, lane = threadIdx.x & 31;
    if (lane == 0) sh[wave] = v;
    __syncthreads();
    if (wave == 0) {
        float t = (lane < 8) ? sh[lane] : 0.f;
#pragma unroll
        for (int o = 4; o > 0; o >>= 1) t += __shfl_xor(t, o);
        if (lane == 0) sh[0] = t;
    }
    __syncthreads();
    float r = sh[0];
    __syncthreads();
    return r;
}

__global__ __launch_bounds__(256) void ln_bf16_k(const float* __restrict__ x,
                                                 const float* __restrict__ g,
                                                 const float* __restrict__ be,
                                                 __bf16* __restrict__ out) {
    __shared__ float sh[8];
    size_t row = blockIdx.x;
    const float* xr = x + row * D_MODEL;
    float v[4]; float s = 0.f;
#pragma unroll
    for (int i = 0; i < 4; ++i) { v[i] = xr[threadIdx.x + i * 256]; s += v[i]; }
    s = block_sum256(s, sh);
    float mu = s * (1.f / D_MODEL);
    float s2 = 0.f;
#pragma unroll
    for (int i = 0; i < 4; ++i) { float d = v[i] - mu; s2 += d * d; }
    s2 = block_sum256(s2, sh);
    float inv = rsqrtf(s2 * (1.f / D_MODEL) + 1e-5f);
#pragma unroll
    for (int i = 0; i < 4; ++i) {
        int c = threadIdx.x + i * 256;
        out[row * D_MODEL + c] = f2bf((v[i] - mu) * inv * g[c] + be[c]);
    }
}

// ---------------------------------------------------------------------------
// bf16 GEMM:  C[M,N] = A[M,K] @ B[N,K]^T + bias (+relu) (+fp32 residual)
// 128x128x32 block tile, double-buffered LDS, 256 threads = 8 waves arranged
// 4(M) x 2(N); each wave owns a 32x64 register tile = 8 WMMA per k-step.
// Global loads for k+32 are staged in registers across the WMMA block.
// ---------------------------------------------------------------------------
template <bool RELU, bool RES, bool OUTBF>
__global__ __launch_bounds__(256) void gemm_bf16_k(
    const __bf16* __restrict__ A, const __bf16* __restrict__ B,
    const float* __restrict__ bias, const float* __restrict__ resid,
    float* __restrict__ Cf, __bf16* __restrict__ Cb, int M, int N, int K) {
    __shared__ __bf16 As[2][128 * 40];
    __shared__ __bf16 Bs[2][128 * 40];
    const int tid = threadIdx.x;
    const int lane = tid & 31, wave = tid >> 5;
    const int wm = wave & 3, wn = wave >> 2;      // 4 M-strips x 2 N-strips
    const int m0 = blockIdx.y * 128, n0 = blockIdx.x * 128;
    const int half = lane >> 4, l15 = lane & 15;
    const int r0 = tid >> 2, r1 = r0 + 64, kc = (tid & 3) * 8;  // load chunks

    // ---- preload k0 = 0 tile into buffer 0 ----
    v8bf a_s0 = *(const v8bf*)&A[(size_t)(m0 + r0) * K + kc];
    v8bf a_s1 = *(const v8bf*)&A[(size_t)(m0 + r1) * K + kc];
    v8bf b_s0 = *(const v8bf*)&B[(size_t)(n0 + r0) * K + kc];
    v8bf b_s1 = *(const v8bf*)&B[(size_t)(n0 + r1) * K + kc];
    *(v8bf*)&As[0][r0 * 40 + kc] = a_s0;
    *(v8bf*)&As[0][r1 * 40 + kc] = a_s1;
    *(v8bf*)&Bs[0][r0 * 40 + kc] = b_s0;
    *(v8bf*)&Bs[0][r1 * 40 + kc] = b_s1;
    __syncthreads();

    v8f acc[2][4] = {};
    int buf = 0;
    for (int k0 = 0; k0 < K; k0 += 32) {
        const bool has_next = (k0 + 32) < K;
        if (has_next) {   // stage next k-slice in registers (overlaps WMMA)
            a_s0 = *(const v8bf*)&A[(size_t)(m0 + r0) * K + k0 + 32 + kc];
            a_s1 = *(const v8bf*)&A[(size_t)(m0 + r1) * K + k0 + 32 + kc];
            b_s0 = *(const v8bf*)&B[(size_t)(n0 + r0) * K + k0 + 32 + kc];
            b_s1 = *(const v8bf*)&B[(size_t)(n0 + r1) * K + k0 + 32 + kc];
            if (k0 + 64 < K) {  // global_prefetch_b8 two slices ahead
                __builtin_prefetch(&A[(size_t)(m0 + r0) * K + k0 + 64 + kc], 0, 1);
                __builtin_prefetch(&B[(size_t)(n0 + r0) * K + k0 + 64 + kc], 0, 1);
            }
        }

        // ---- 8 WMMAs on current buffer ----
        v16bf a0 = ld_fragA(&As[buf][(wm * 32 + l15) * 40], half);
        v16bf a1 = ld_fragA(&As[buf][(wm * 32 + 16 + l15) * 40], half);
#pragma unroll
        for (int t = 0; t < 4; ++t) {
            v16bf bf = ld_fragB(&Bs[buf][(wn * 64 + t * 16 + l15) * 40], half);
            acc[0][t] = wmma_bf16(a0, bf, acc[0][t]);
            acc[1][t] = wmma_bf16(a1, bf, acc[1][t]);
        }

        if (has_next) {
            const int nb = buf ^ 1;
            *(v8bf*)&As[nb][r0 * 40 + kc] = a_s0;
            *(v8bf*)&As[nb][r1 * 40 + kc] = a_s1;
            *(v8bf*)&Bs[nb][r0 * 40 + kc] = b_s0;
            *(v8bf*)&Bs[nb][r1 * 40 + kc] = b_s1;
        }
        __syncthreads();
        buf ^= 1;
    }

    // ---- epilogue: C layout (ISA 7.12.2): row = half*8+r, col = l15 ----
#pragma unroll
    for (int u = 0; u < 2; ++u) {
#pragma unroll
        for (int t = 0; t < 4; ++t) {
            const int col = n0 + wn * 64 + t * 16 + l15;
            const float bi = bias[col];
#pragma unroll
            for (int r = 0; r < 8; ++r) {
                size_t row = m0 + wm * 32 + u * 16 + half * 8 + r;
                float v = acc[u][t][r] + bi;
                if (RELU) v = fmaxf(v, 0.f);
                if (RES)  v += resid[row * N + col];
                if (OUTBF) Cb[row * N + col] = f2bf(v);
                else       Cf[row * N + col] = v;
            }
        }
    }
}

// ---------------------------------------------------------------------------
// RoPE + QKV split:  qkv fp32 [NTOK,3072] -> bf16 Q/K (rope'd) and V,
// layout [B*H, SEQ, 64].  One block per token.
// ---------------------------------------------------------------------------
__global__ __launch_bounds__(256) void rope_split_k(const float* __restrict__ qkv,
                                                    __bf16* __restrict__ Qb,
                                                    __bf16* __restrict__ Kb,
                                                    __bf16* __restrict__ Vb) {
    int t = blockIdx.x;             // token index
    int s = t & (SEQ - 1);
    int b = t >> 11;
    const float* row = qkv + (size_t)t * 3 * D_MODEL;
    for (int i = threadIdx.x; i < N_HEADS * 32; i += 256) {
        int h = i >> 5, p = i & 31;
        float inv = __powf(10000.f, -(float)(2 * p) * (1.f / HEAD_DIM));
        float ang = (float)s * inv;
        float sn, cs; __sincosf(ang, &sn, &cs);
        float qe = row[h * 64 + 2 * p],        qo = row[h * 64 + 2 * p + 1];
        float ke = row[1024 + h * 64 + 2 * p], ko = row[1024 + h * 64 + 2 * p + 1];
        size_t o = ((size_t)(b * N_HEADS + h) * SEQ + s) * HEAD_DIM;
        Qb[o + 2 * p]     = f2bf(qe * cs - qo * sn);
        Qb[o + 2 * p + 1] = f2bf(qe * sn + qo * cs);
        Kb[o + 2 * p]     = f2bf(ke * cs - ko * sn);
        Kb[o + 2 * p + 1] = f2bf(ke * sn + ko * cs);
    }
    for (int i = threadIdx.x; i < D_MODEL; i += 256) {
        int h = i >> 6, d = i & 63;
        Vb[((size_t)(b * N_HEADS + h) * SEQ + s) * HEAD_DIM + d] = f2bf(row[2048 + i]);
    }
}

// ---------------------------------------------------------------------------
// Flash attention (causal).  grid = (S/64 q-tiles, B*H).  128 threads = 4
// waves, wave w owns q-rows w*16..w*16+15.  QK^T and PV via bf16 WMMA,
// online softmax in registers (v_exp_f32 trans ops co-execute with WMMA).
// ---------------------------------------------------------------------------
__global__ __launch_bounds__(128) void flash_attn_k(const __bf16* __restrict__ Q,
                                                    const __bf16* __restrict__ K,
                                                    const __bf16* __restrict__ V,
                                                    __bf16* __restrict__ ctx) {
    __shared__ __bf16 Ks[64 * 72];       // [key][d]
    __shared__ __bf16 Vt[64 * 72];       // [d][key]  (transposed for B-operand)
    __shared__ __bf16 Ps[4 * 16 * 72];   // per-wave P tile [16 rows][64 keys]
    const int tid = threadIdx.x;
    const int lane = tid & 31, wave = tid >> 5;
    const int half = lane >> 4, l15 = lane & 15;
    const int qt = blockIdx.x, bh = blockIdx.y;
    const int b = bh >> 4, h = bh & 15;
    const int q0 = qt * 64;
    const size_t seq_base = (size_t)bh * SEQ;

    // Q fragments (A-operand) straight from global, rope already applied
    const __bf16* qrow = Q + (seq_base + q0 + wave * 16 + l15) * HEAD_DIM;
    v16bf qf0 = ld_fragA(qrow, half);
    v16bf qf1 = ld_fragA(qrow + 32, half);

    v8f acc_o[4] = {v8f{}, v8f{}, v8f{}, v8f{}};
    float m_run[8], l_run[8];
#pragma unroll
    for (int r = 0; r < 8; ++r) { m_run[r] = -3.0e38f; l_run[r] = 0.f; }

    __bf16* ps = &Ps[wave * 16 * 72];

    for (int j = 0; j <= qt; ++j) {
        __syncthreads();
        // cooperative load of K tile + transposed V tile
#pragma unroll
        for (int c = 0; c < 4; ++c) {
            int idx = tid + c * 128;          // 0..511 chunks of 8 bf16
            int row = idx >> 3, kk = (idx & 7) * 8;
            size_t g = (seq_base + (size_t)j * 64 + row) * HEAD_DIM + kk;
            *(v8bf*)&Ks[row * 72 + kk] = *(const v8bf*)&K[g];
            v8bf vv = *(const v8bf*)&V[g];
#pragma unroll
            for (int e = 0; e < 8; ++e) Vt[(kk + e) * 72 + row] = vv[e];
        }
        __syncthreads();

        // S = Q K^T  (16x64 per wave: 4 n-tiles x 2 k-steps of 32)
        v8f sacc[4] = {v8f{}, v8f{}, v8f{}, v8f{}};
#pragma unroll
        for (int ks = 0; ks < 2; ++ks) {
            v16bf a = ks ? qf1 : qf0;
#pragma unroll
            for (int t = 0; t < 4; ++t) {
                v16bf bf = ld_fragB(&Ks[(t * 16 + l15) * 72 + ks * 32], half);
                sacc[t] = wmma_bf16(a, bf, sacc[t]);
            }
        }

        // causal mask + online softmax (rows m = half*8 + r, cols n = l15)
        float p[4][8];
#pragma unroll
        for (int r = 0; r < 8; ++r) {
            int qg = q0 + wave * 16 + half * 8 + r;
            float rm = -3.0e38f;
#pragma unroll
            for (int t = 0; t < 4; ++t) {
                float sv = sacc[t][r] * 0.125f;         // 1/sqrt(64)
                int kg = j * 64 + t * 16 + l15;
                if (kg > qg) sv = -3.0e38f;
                p[t][r] = sv;
                rm = fmaxf(rm, sv);
            }
#pragma unroll
            for (int o = 1; o < 16; o <<= 1) rm = fmaxf(rm, __shfl_xor(rm, o));
            float mn = fmaxf(m_run[r], rm);
            float alpha = __expf(m_run[r] - mn);
            float rs = 0.f;
#pragma unroll
            for (int t = 0; t < 4; ++t) { float e = __expf(p[t][r] - mn); p[t][r] = e; rs += e; }
#pragma unroll
            for (int o = 1; o < 16; o <<= 1) rs += __shfl_xor(rs, o);
            l_run[r] = l_run[r] * alpha + rs;
            m_run[r] = mn;
#pragma unroll
            for (int t = 0; t < 4; ++t) acc_o[t][r] *= alpha;
        }

        // stage P (C-layout) into per-wave LDS, re-read as A-operand fragments
#pragma unroll
        for (int r = 0; r < 8; ++r)
#pragma unroll
            for (int t = 0; t < 4; ++t)
                ps[(half * 8 + r) * 72 + t * 16 + l15] = f2bf(p[t][r]);
        asm volatile("s_wait_dscnt 0x0" ::: "memory");   // CDNA5 split DS counter
        __builtin_amdgcn_wave_barrier();

        // O += P V   (k = key dim, 2 steps of 32)
#pragma unroll
        for (int ks = 0; ks < 2; ++ks) {
            v16bf pa = ld_fragA(&ps[l15 * 72 + ks * 32], half);
#pragma unroll
            for (int t = 0; t < 4; ++t) {
                v16bf vb = ld_fragB(&Vt[(t * 16 + l15) * 72 + ks * 32], half);
                acc_o[t] = wmma_bf16(pa, vb, acc_o[t]);
            }
        }
    }

    // write ctx [NTOK, D_MODEL] bf16
#pragma unroll
    for (int r = 0; r < 8; ++r) {
        float inv = 1.f / l_run[r];
        size_t row = (size_t)b * SEQ + q0 + wave * 16 + half * 8 + r;
#pragma unroll
        for (int t = 0; t < 4; ++t)
            ctx[row * D_MODEL + h * HEAD_DIM + t * 16 + l15] = f2bf(acc_o[t][r] * inv);
    }
}

// ---------------------------------------------------------------------------
// Launcher
// ---------------------------------------------------------------------------
extern "C" void kernel_launch(void* const* d_in, const int* in_sizes, int n_in,
                              void* d_out, int out_size, void* d_ws, size_t ws_size,
                              hipStream_t stream) {
    (void)in_sizes; (void)n_in; (void)out_size; (void)ws_size;
    const float* src   = (const float*)d_in[0];
    const float* ln1_g = (const float*)d_in[1];
    const float* ln1_b = (const float*)d_in[2];
    const float* wqkv  = (const float*)d_in[3];
    const float* bqkv  = (const float*)d_in[4];
    const float* wo    = (const float*)d_in[5];
    const float* bo    = (const float*)d_in[6];
    const float* ln2_g = (const float*)d_in[7];
    const float* ln2_b = (const float*)d_in[8];
    const float* w1    = (const float*)d_in[9];
    const float* b1    = (const float*)d_in[10];
    const float* w2    = (const float*)d_in[11];
    const float* b2    = (const float*)d_in[12];
    float* out = (float*)d_out;

    char* base = (char*)d_ws;
    size_t off = 0;
    auto alloc = [&](size_t bytes) -> void* {
        void* p = base + off;
        off += (bytes + 255) & ~(size_t)255;
        return p;
    };
    __bf16* wqkv_bf = (__bf16*)alloc((size_t)3 * D_MODEL * D_MODEL * 2);
    __bf16* wo_bf   = (__bf16*)alloc((size_t)D_MODEL * D_MODEL * 2);
    __bf16* w1_bf   = (__bf16*)alloc((size_t)D_FF * D_MODEL * 2);
    __bf16* w2_bf   = (__bf16*)alloc((size_t)D_MODEL * D_FF * 2);
    __bf16* xln_bf  = (__bf16*)alloc((size_t)NTOK * D_MODEL * 2);
    float*  qkv_f   = (float* )alloc((size_t)NTOK * 3 * D_MODEL * 4);
    __bf16* q_bf    = (__bf16*)alloc((size_t)NTOK * D_MODEL * 2);
    __bf16* k_bf    = (__bf16*)alloc((size_t)NTOK * D_MODEL * 2);
    __bf16* v_bf    = (__bf16*)alloc((size_t)NTOK * D_MODEL * 2);
    __bf16* ctx_bf  = (__bf16*)alloc((size_t)NTOK * D_MODEL * 2);
    float*  src2_f  = (float* )alloc((size_t)NTOK * D_MODEL * 4);
    __bf16* ln2_bf  = (__bf16*)alloc((size_t)NTOK * D_MODEL * 2);
    __bf16* h_bf    = (__bf16*)alloc((size_t)NTOK * D_FF * 2);

    auto cvt = [&](const float* s, __bf16* d, int n) {
        to_bf16_k<<<(n + 255) / 256, 256, 0, stream>>>(s, d, n);
    };
    cvt(wqkv, wqkv_bf, 3 * D_MODEL * D_MODEL);
    cvt(wo,   wo_bf,   D_MODEL * D_MODEL);
    cvt(w1,   w1_bf,   D_FF * D_MODEL);
    cvt(w2,   w2_bf,   D_MODEL * D_FF);

    // LN1 -> bf16
    ln_bf16_k<<<NTOK, 256, 0, stream>>>(src, ln1_g, ln1_b, xln_bf);

    // QKV GEMM: [4096,1024] x [3072,1024]^T -> fp32
    gemm_bf16_k<false, false, false><<<dim3(3 * D_MODEL / 128, NTOK / 128), 256, 0, stream>>>(
        xln_bf, wqkv_bf, bqkv, nullptr, qkv_f, nullptr, NTOK, 3 * D_MODEL, D_MODEL);

    // RoPE + split into per-head bf16 Q/K/V
    rope_split_k<<<NTOK, 256, 0, stream>>>(qkv_f, q_bf, k_bf, v_bf);

    // Causal flash attention -> ctx bf16
    flash_attn_k<<<dim3(SEQ / 64, BATCH * N_HEADS), 128, 0, stream>>>(q_bf, k_bf, v_bf, ctx_bf);

    // Wo GEMM + bias + residual(src) -> src2 fp32
    gemm_bf16_k<false, true, false><<<dim3(D_MODEL / 128, NTOK / 128), 256, 0, stream>>>(
        ctx_bf, wo_bf, bo, src, src2_f, nullptr, NTOK, D_MODEL, D_MODEL);

    // LN2 -> bf16
    ln_bf16_k<<<NTOK, 256, 0, stream>>>(src2_f, ln2_g, ln2_b, ln2_bf);

    // W1 GEMM + bias + ReLU -> bf16
    gemm_bf16_k<true, false, true><<<dim3(D_FF / 128, NTOK / 128), 256, 0, stream>>>(
        ln2_bf, w1_bf, b1, nullptr, nullptr, h_bf, NTOK, D_FF, D_MODEL);

    // W2 GEMM + bias + residual(src2) -> d_out fp32
    gemm_bf16_k<false, true, false><<<dim3(D_MODEL / 128, NTOK / 128), 256, 0, stream>>>(
        h_bf, w2_bf, b2, src2_f, out, nullptr, NTOK, D_MODEL, D_FF);
}
